// QuantLinear_61847529063064
// MI455X (gfx1250) — matrix operand order repaired and checked by reading, not compiled, not run on hardware
//
#include <hip/hip_runtime.h>

typedef __attribute__((ext_vector_type(16))) _Float16 v16h;
typedef __attribute__((ext_vector_type(8)))  _Float16 h8;
typedef __attribute__((ext_vector_type(4)))  _Float16 h4;
typedef __attribute__((ext_vector_type(8)))  float    v8f;
typedef __attribute__((ext_vector_type(4)))  float    f4;
typedef unsigned int u32x4 __attribute__((ext_vector_type(4)));
typedef int          i32x4 __attribute__((ext_vector_type(4)));
typedef int          i32x8 __attribute__((ext_vector_type(8)));

#define K_IN   4096
#define N_OUT  11008
#define NWORDS (N_OUT / 8)   // 1376 packed words per qweight row
#define BM     128
#define BN     128
#define BK     32
#define NKT    (K_IN / BK)   // 128 K-steps
#define LDA    40            // padded f16 stride (80B = 5*16B -> aligned + conflict-free)
#define LDB    40

#if defined(__has_builtin)
#if __has_builtin(__builtin_amdgcn_tensor_load_to_lds) && \
    __has_builtin(__builtin_amdgcn_s_wait_tensorcnt)
#define HAS_TDM 1
#endif
#endif
#ifndef HAS_TDM
#define HAS_TDM 0
#endif

__device__ __forceinline__ unsigned lds_off(const void* p) {
    // Generic pointers into LDS carry the LDS byte offset in the low 32 bits
    // (flat aperture mapping).
    return (unsigned)(unsigned long long)p;
}

#if HAS_TDM
// Issue one TDM 2D tile load (4-byte elements): tile (tile_d0 x tile_d1) of a
// (tensor_d0 x tensor_d1) row-major tensor with row stride stride0 (elements).
__device__ __forceinline__ void tdm_load_2d(unsigned lds_addr, unsigned long long gaddr,
                                            unsigned tensor_d0, unsigned tensor_d1,
                                            unsigned tile_d0, unsigned tile_d1,
                                            unsigned stride0) {
    // D# group 0: count=1 | lds_addr | global_addr[56:0] | type=2 (bits 127:126)
    u32x4 g0 = { 1u,
                 lds_addr,
                 (unsigned)gaddr,
                 (unsigned)((gaddr >> 32) & 0x01FFFFFFull) | 0x80000000u };
    // D# group 1: data_size=2 (4B) @ bits17:16 | tensor_dim0 @79:48 |
    //             tensor_dim1 @111:80 | tile_dim0 @127:112 | tile_dim1 @143:128 |
    //             tensor_dim0_stride @207:160
    i32x8 g1 = { (int)0x00020000,
                 (int)((tensor_d0 & 0xFFFFu) << 16),
                 (int)(((tensor_d0 >> 16) & 0xFFFFu) | ((tensor_d1 & 0xFFFFu) << 16)),
                 (int)(((tensor_d1 >> 16) & 0xFFFFu) | ((tile_d0 & 0xFFFFu) << 16)),
                 (int)(tile_d1 & 0xFFFFu),
                 (int)stride0,
                 0, 0 };
    i32x4 zz = { 0, 0, 0, 0 };
#if __clang_major__ >= 23
    i32x8 z8 = { 0, 0, 0, 0, 0, 0, 0, 0 };
    __builtin_amdgcn_tensor_load_to_lds(g0, g1, zz, zz, z8, 0);
#else
    __builtin_amdgcn_tensor_load_to_lds(g0, g1, zz, zz, 0);
#endif
}
#endif

__global__ __launch_bounds__(256, 2)
void qlinear_w4a16_wmma(const float* __restrict__ x,
                        const unsigned int* __restrict__ qweight,
                        const unsigned int* __restrict__ qzeros,
                        const float* __restrict__ scales,
                        const float* __restrict__ bias,
                        float* __restrict__ out,
                        int Mtotal)
{
    __shared__ _Float16 As[BM * LDA];   // x tile, f16, row-major [m][k]
    __shared__ _Float16 Bt[BN * LDB];   // dequant W tile, f16, transposed [n][k]
    __shared__ float    zsh[BN];        // per-column zero (current group)
    __shared__ float    ssh[BN];        // per-column scale (current group)
#if HAS_TDM
    __shared__ float        LdsX[2][BM * BK];   // TDM staging: x tile (f32, dense [m][k])
    __shared__ unsigned int LdsQ[2][BK * 16];   // TDM staging: qweight tile [k][word]
#endif

    const int tid   = threadIdx.x;
    const int lane  = tid & 31;
    const int wave  = tid >> 5;
    const int waveM = wave >> 2;        // 0..1
    const int waveN = wave & 3;         // 0..3
    const int m0    = blockIdx.y * BM;
    const int n0    = blockIdx.x * BN;
    const int nw0   = n0 >> 3;

    const int row   = lane & 15;        // A: M row / B: N col / D: N col
    const int khalf = lane >> 4;        // 0 or 1

    v8f acc[4][2] = {};

#if HAS_TDM
    // Prologue: wave 0 kicks off the DMA for tile 0 into staging buffer 0.
    if (wave == 0) {
        tdm_load_2d(lds_off(&LdsX[0][0]),
                    (unsigned long long)(x + (size_t)m0 * K_IN),
                    K_IN, (unsigned)Mtotal, BK, BM, K_IN);
        tdm_load_2d(lds_off(&LdsQ[0][0]),
                    (unsigned long long)(qweight + nw0),
                    NWORDS, K_IN, 16, BK, NWORDS);
    }
#endif

    #pragma unroll 1
    for (int kt = 0; kt < NKT; ++kt) {
        const int k0  = kt * BK;
        const int cur = kt & 1;

        // Refresh per-group dequant params every 4 K-steps (GROUP_SIZE=128=4*BK)
        if ((kt & 3) == 0 && tid < BN) {
            const int g = k0 >> 7;
            const unsigned int zq = qzeros[(size_t)g * NWORDS + nw0 + (tid >> 3)];
            zsh[tid] = (float)((zq >> ((tid & 7) * 4)) & 0xFu);
            ssh[tid] = scales[(size_t)g * N_OUT + n0 + tid];
        }

#if HAS_TDM
        // ---- Software pipeline: issue DMA for tile kt+1 into the alternate
        //      buffer, then wait until only those 2 newest loads remain in
        //      flight (TDM completes in-order per wave => tile kt is ready).
        //      Last iteration issues a harmless re-fetch of tile 0 (nothing
        //      reads it; s_endpgm's implicit wait-idle drains it). ----
        if (wave == 0) {
            const int kn = ((kt + 1) & (NKT - 1)) * BK;
            tdm_load_2d(lds_off(&LdsX[cur ^ 1][0]),
                        (unsigned long long)(x + (size_t)m0 * K_IN + kn),
                        K_IN, (unsigned)Mtotal, BK, BM, K_IN);
            tdm_load_2d(lds_off(&LdsQ[cur ^ 1][0]),
                        (unsigned long long)(qweight + (size_t)kn * NWORDS + nw0),
                        NWORDS, K_IN, 16, BK, NWORDS);
            __builtin_amdgcn_s_wait_tensorcnt(2);
        }
        __syncthreads();   // publish tile kt + zs/ss refresh to all waves
#else
        if ((kt & 3) == 0) __syncthreads();
#endif

        // ---- Stage A: x tile f32 -> f16 into LDS (row-major, padded) ----
        {
            const int c4 = tid & 7;                      // which float4 in the 32-wide row
            #pragma unroll
            for (int i = 0; i < 4; ++i) {
                const int r = (tid >> 3) + 32 * i;       // 0..127
#if HAS_TDM
                const f4 v = *(const f4*)(&LdsX[cur][0] + r * BK + c4 * 4);
#else
                const f4 v = *(const f4*)(x + (size_t)(m0 + r) * K_IN + k0 + c4 * 4);
#endif
                h4 h;
                h.x = (_Float16)v.x; h.y = (_Float16)v.y;
                h.z = (_Float16)v.z; h.w = (_Float16)v.w;
                *(h4*)(As + r * LDA + c4 * 4) = h;       // ds_store_b64
            }
        }

        // ---- Stage B: dequant int4 -> f16, store K-transposed ----
        #pragma unroll
        for (int i = 0; i < 2; ++i) {
            const int w  = tid + 256 * i;                // 0..511
            const int kk = w >> 4;                       // 0..31  (K within tile)
            const int n8 = w & 15;                       // packed word within row
#if HAS_TDM
            const unsigned int qwv = LdsQ[cur][kk * 16 + n8];
#else
            const unsigned int qwv = qweight[(size_t)(k0 + kk) * NWORDS + nw0 + n8];
#endif
            #pragma unroll
            for (int j = 0; j < 8; ++j) {
                const int n = n8 * 8 + j;
                const float wv = ((float)((qwv >> (4 * j)) & 0xFu) - zsh[n]) * ssh[n];
                Bt[n * LDB + kk] = (_Float16)wv;         // ds_store_b16
            }
        }
        __syncthreads();

        // ---- Fragments (layouts per CDNA5 ISA 7.12.2) ----
        v16h a[4];
        #pragma unroll
        for (int i = 0; i < 4; ++i) {
            const int ar = waveM * 64 + i * 16 + row;
            const h8 lo = *(const h8*)(As + ar * LDA + khalf * 8);        // K = kh*8 + 0..7
            const h8 hi = *(const h8*)(As + ar * LDA + khalf * 8 + 16);   // K = kh*8 + 16..23
            #pragma unroll
            for (int e = 0; e < 8; ++e) { a[i][e] = lo[e]; a[i][e + 8] = hi[e]; }
        }
        v16h b[2];
        #pragma unroll
        for (int j = 0; j < 2; ++j) {
            const int bn = waveN * 32 + j * 16 + row;
            const h8 lo = *(const h8*)(Bt + bn * LDB + khalf * 8);        // K = kh*8 + 0..7
            const h8 hi = *(const h8*)(Bt + bn * LDB + khalf * 8 + 16);   // K = kh*8 + 16..23
            #pragma unroll
            for (int e = 0; e < 8; ++e) { b[j][e] = lo[e]; b[j][e + 8] = hi[e]; }
        }

        #pragma unroll
        for (int i = 0; i < 4; ++i)
            #pragma unroll
            for (int j = 0; j < 2; ++j)
                acc[i][j] = __builtin_amdgcn_wmma_f32_16x16x32_f16(
                    false, a[i], false, b[j], (short)0, acc[i][j], false, false);

        __syncthreads();
    }

    // ---- Epilogue: bias add + store (C/D layout: VGPR r -> M = r + 8*khalf) ----
    #pragma unroll
    for (int j = 0; j < 2; ++j) {
        const int n = n0 + waveN * 32 + j * 16 + row;
        const float bv = bias[n];
        #pragma unroll
        for (int i = 0; i < 4; ++i) {
            const int mbase = m0 + waveM * 64 + i * 16 + khalf * 8;
            #pragma unroll
            for (int r = 0; r < 8; ++r)
                out[(size_t)(mbase + r) * N_OUT + n] = acc[i][j][r] + bv;
        }
    }
}

extern "C" void kernel_launch(void* const* d_in, const int* in_sizes, int n_in,
                              void* d_out, int out_size, void* d_ws, size_t ws_size,
                              hipStream_t stream) {
    const float*        xp = (const float*)d_in[0];
    const unsigned int* qw = (const unsigned int*)d_in[1];
    const unsigned int* qz = (const unsigned int*)d_in[2];
    const float*        sc = (const float*)d_in[3];
    const float*        bi = (const float*)d_in[4];
    float*             outp = (float*)d_out;

    const int M = in_sizes[0] / K_IN;                 // 2048 tokens
    dim3 grid(N_OUT / BN, M / BM);                    // (86, 16)
    qlinear_w4a16_wmma<<<grid, 256, 0, stream>>>(xp, qw, qz, sc, bi, outp, M);
}
